// WeightOnlyInt8Linear_5952824672395
// MI455X (gfx1250) — compile-verified, matrix operand order checked by reading, not run on hardware
//
#include <hip/hip_runtime.h>

typedef __attribute__((ext_vector_type(16))) _Float16    v16h;
typedef __attribute__((ext_vector_type(8)))  _Float16    v8h;
typedef __attribute__((ext_vector_type(4)))  _Float16    v4h;
typedef __attribute__((ext_vector_type(8)))  float       v8f;
typedef __attribute__((ext_vector_type(4)))  float       v4f;
typedef __attribute__((ext_vector_type(8)))  signed char v8c;
typedef __attribute__((ext_vector_type(16))) signed char v16c;

// Problem constants: M = B*S = 8192, K = 4096, N = 11008
constexpr int Mdim = 4 * 2048;
constexpr int Ndim = 11008;
constexpr int Kdim = 4096;

constexpr int TM = 128;        // block M tile
constexpr int TN = 256;        // block N tile
constexpr int TK = 64;         // K chunk per LDS stage (2 WMMA k-steps)
constexpr int LDSS = TK + 8;   // padded LDS stride in halfs (144B rows, conflict-free)

// ---------------- one-time precision conversion kernels ----------------

__global__ __launch_bounds__(256)
void convert_x_f16(const float* __restrict__ x, _Float16* __restrict__ xh) {
    size_t i = ((size_t)blockIdx.x * 256 + threadIdx.x) * 8;
    v4f a = *(const v4f*)(x + i);
    v4f b = *(const v4f*)(x + i + 4);
    v8h h;
#pragma unroll
    for (int j = 0; j < 4; ++j) { h[j] = (_Float16)a[j]; h[4 + j] = (_Float16)b[j]; }
    *(v8h*)(xh + i) = h;
}

__global__ __launch_bounds__(256)
void convert_w_f16(const signed char* __restrict__ w, _Float16* __restrict__ wh) {
    size_t i = ((size_t)blockIdx.x * 256 + threadIdx.x) * 16;
    v16c c = *(const v16c*)(w + i);
    v8h h0, h1;
#pragma unroll
    for (int j = 0; j < 8; ++j) {
        h0[j] = (_Float16)(short)c[j];       // int8 is exact in f16
        h1[j] = (_Float16)(short)c[8 + j];
    }
    *(v8h*)(wh + i)     = h0;
    *(v8h*)(wh + i + 8) = h1;
}

// ---------------- WMMA GEMM ----------------
// PRE=true : A/B already f16 in workspace -> zero-VALU data movement
// PRE=false: fused conversion fallback (used only if workspace too small)

template <bool PRE>
__global__ __launch_bounds__(256)
void gemm_f16wmma(const _Float16* __restrict__ xh, const _Float16* __restrict__ wh,
                  const float* __restrict__ xf, const signed char* __restrict__ w8,
                  const float* __restrict__ scales, float* __restrict__ out)
{
    __shared__ _Float16 lds_a[TM * LDSS];   // 18.4 KB
    __shared__ _Float16 lds_b[TN * LDSS];   // 36.9 KB

    const int tid  = threadIdx.x;
    const int lane = tid & 31;
    const int wave = tid >> 5;
    const int wm   = wave >> 2;            // 0..1 : 64-row slab
    const int wn   = wave & 3;             // 0..3 : 64-col slab

    const int bm = blockIdx.y;
    const int bn = blockIdx.x;

    v8f acc[4][4];
#pragma unroll
    for (int mt = 0; mt < 4; ++mt)
#pragma unroll
        for (int nt = 0; nt < 4; ++nt)
            acc[mt][nt] = (v8f){0.f, 0.f, 0.f, 0.f, 0.f, 0.f, 0.f, 0.f};

    // prefetch registers (only one set survives per instantiation)
    v8h pa_h[4], pb_h[8];     // PRE path: raw f16 chunks
    v4f pa_f[8];              // fallback: f32 x
    v8c pb_c[8];              // fallback: int8 w

    const _Float16*    xh_b = xh + (size_t)(bm * TM) * Kdim;
    const _Float16*    wh_b = wh + (size_t)(bn * TN) * Kdim;
    const float*       xf_b = xf + (size_t)(bm * TM) * Kdim;
    const signed char* w8_b = w8 + (size_t)(bn * TN) * Kdim;

    auto load_stage = [&](int k0) {
        if constexpr (PRE) {
#pragma unroll
            for (int j = 0; j < 4; ++j) {            // A: 1024 8-half chunks
                int idx = tid + 256 * j;
                int r = idx >> 3, c = (idx & 7) * 8;
                pa_h[j] = *(const v8h*)(xh_b + (size_t)r * Kdim + k0 + c);
            }
#pragma unroll
            for (int j = 0; j < 8; ++j) {            // B: 2048 8-half chunks
                int idx = tid + 256 * j;
                int r = idx >> 3, c = (idx & 7) * 8;
                pb_h[j] = *(const v8h*)(wh_b + (size_t)r * Kdim + k0 + c);
            }
        } else {
#pragma unroll
            for (int j = 0; j < 8; ++j) {            // A: 2048 float4 chunks
                int idx = tid + 256 * j;
                int r = idx >> 4, c = (idx & 15) * 4;
                pa_f[j] = *(const v4f*)(xf_b + (size_t)r * Kdim + k0 + c);
            }
#pragma unroll
            for (int j = 0; j < 8; ++j) {            // B: 2048 8-byte chunks
                int idx = tid + 256 * j;
                int r = idx >> 3, c = (idx & 7) * 8;
                pb_c[j] = *(const v8c*)(w8_b + (size_t)r * Kdim + k0 + c);
            }
        }
    };

    auto store_stage = [&]() {
        if constexpr (PRE) {
#pragma unroll
            for (int j = 0; j < 4; ++j) {
                int idx = tid + 256 * j;
                int r = idx >> 3, c = (idx & 7) * 8;
                *(v8h*)(&lds_a[r * LDSS + c]) = pa_h[j];
            }
#pragma unroll
            for (int j = 0; j < 8; ++j) {
                int idx = tid + 256 * j;
                int r = idx >> 3, c = (idx & 7) * 8;
                *(v8h*)(&lds_b[r * LDSS + c]) = pb_h[j];
            }
        } else {
#pragma unroll
            for (int j = 0; j < 8; ++j) {
                int idx = tid + 256 * j;
                int r = idx >> 4, c = (idx & 15) * 4;
                v4h h;
#pragma unroll
                for (int i = 0; i < 4; ++i) h[i] = (_Float16)pa_f[j][i];
                *(v4h*)(&lds_a[r * LDSS + c]) = h;
            }
#pragma unroll
            for (int j = 0; j < 8; ++j) {
                int idx = tid + 256 * j;
                int r = idx >> 3, c = (idx & 7) * 8;
                v8h h;
#pragma unroll
                for (int i = 0; i < 8; ++i) h[i] = (_Float16)(short)pb_c[j][i];
                *(v8h*)(&lds_b[r * LDSS + c]) = h;
            }
        }
    };

    load_stage(0);

    for (int k0 = 0; k0 < Kdim; k0 += TK) {
        store_stage();
        __syncthreads();

        if (k0 + TK < Kdim) load_stage(k0 + TK);   // overlap next global fetch

#pragma unroll
        for (int ks = 0; ks < 2; ++ks) {
            // ISA A/B f16 layout: lanes 0-15: K base+0..7 / +16..23;
            //                     lanes 16-31: K base+8..15 / +24..31
            const int kb = ks * 32 + ((lane >> 4) << 3);

            v16h af[4];
#pragma unroll
            for (int mt = 0; mt < 4; ++mt) {
                int m = wm * 64 + mt * 16 + (lane & 15);
                v8h lo = *(const v8h*)(&lds_a[m * LDSS + kb]);
                v8h hi = *(const v8h*)(&lds_a[m * LDSS + kb + 16]);
                af[mt] = __builtin_shufflevector(lo, hi,
                    0,1,2,3,4,5,6,7,8,9,10,11,12,13,14,15);
            }
            v16h bf[4];
#pragma unroll
            for (int nt = 0; nt < 4; ++nt) {
                int n = wn * 64 + nt * 16 + (lane & 15);
                v8h lo = *(const v8h*)(&lds_b[n * LDSS + kb]);
                v8h hi = *(const v8h*)(&lds_b[n * LDSS + kb + 16]);
                bf[nt] = __builtin_shufflevector(lo, hi,
                    0,1,2,3,4,5,6,7,8,9,10,11,12,13,14,15);
            }
#pragma unroll
            for (int mt = 0; mt < 4; ++mt)
#pragma unroll
                for (int nt = 0; nt < 4; ++nt)
                    acc[mt][nt] = __builtin_amdgcn_wmma_f32_16x16x32_f16(
                        false, af[mt], false, bf[nt],
                        (short)0, acc[mt][nt], false, false);
        }
        __syncthreads();
    }

    // epilogue: per-output-channel dequant scale, then store
    const int col0 = bn * TN + wn * 64 + (lane & 15);
    const int row0 = bm * TM + wm * 64 + ((lane >> 4) << 3);
#pragma unroll
    for (int nt = 0; nt < 4; ++nt) {
        const int col = col0 + nt * 16;
        const float s = scales[col];
#pragma unroll
        for (int mt = 0; mt < 4; ++mt) {
            const int row = row0 + mt * 16;
#pragma unroll
            for (int i = 0; i < 8; ++i)
                out[(size_t)(row + i) * Ndim + col] = acc[mt][nt][i] * s;
        }
    }
}

// ---------------- launch ----------------

extern "C" void kernel_launch(void* const* d_in, const int* in_sizes, int n_in,
                              void* d_out, int out_size, void* d_ws, size_t ws_size,
                              hipStream_t stream) {
    const float*       x      = (const float*)d_in[0];
    const signed char* weight = (const signed char*)d_in[1];   // int8 [OUT][IN]
    const float*       scales = (const float*)d_in[2];
    float*             out    = (float*)d_out;

    constexpr size_t WS_X = (size_t)Mdim * Kdim * 2;   // 64 MB f16 x
    constexpr size_t WS_W = (size_t)Ndim * Kdim * 2;   // 90 MB f16 w

    dim3 grid(Ndim / TN, Mdim / TM);   // (43, 64)
    dim3 block(256);                   // 8 wave32

    if (ws_size >= WS_X + WS_W) {
        _Float16* xh = (_Float16*)d_ws;
        _Float16* wh = (_Float16*)((char*)d_ws + WS_X);
        convert_x_f16<<<(int)((size_t)Mdim * Kdim / (8 * 256)), 256, 0, stream>>>(x, xh);
        convert_w_f16<<<(int)((size_t)Ndim * Kdim / (16 * 256)), 256, 0, stream>>>(weight, wh);
        gemm_f16wmma<true><<<grid, block, 0, stream>>>(xh, wh, nullptr, nullptr, scales, out);
    } else {
        gemm_f16wmma<false><<<grid, block, 0, stream>>>(nullptr, nullptr, x, weight, scales, out);
    }
}
